// LayerAllocationModule_8160437862927
// MI455X (gfx1250) — matrix operand order repaired and checked by reading, compiled, not run on hardware
//
#include <hip/hip_runtime.h>

// ---------------------------------------------------------------------------
// CDNA5 / gfx1250 fused MLP + softmax-top6 straight-through head.
// B=131072 rows, 256 -> 256 -> 256 -> 24, bf16 WMMA (v_wmma_f32_16x16x32_bf16),
// fp32 accumulate.  One wave == one 16-row tile.  8 waves / 256-thread block.
// ---------------------------------------------------------------------------

typedef __attribute__((ext_vector_type(16))) __bf16        v16bf;
typedef __attribute__((ext_vector_type(8)))  float         v8f;
typedef __attribute__((ext_vector_type(4)))  float         v4f;
typedef __attribute__((ext_vector_type(4)))  unsigned int  v4u;

union Frag {
  v4u   q[2];   // 8 dwords = 16 bf16
  v16bf bf;
};

__device__ __forceinline__ unsigned short f2bf(float f) {
  // fp32 -> bf16 round-to-nearest-even (inputs are finite normals)
  unsigned int u = __float_as_uint(f);
  u += 0x7FFFu + ((u >> 16) & 1u);
  return (unsigned short)(u >> 16);
}
__device__ __forceinline__ unsigned int pk2(float a, float b) {
  return (unsigned int)f2bf(a) | ((unsigned int)f2bf(b) << 16);
}
__device__ __forceinline__ v4u pack8(v4f a, v4f b) {
  v4u r = { pk2(a.x, a.y), pk2(a.z, a.w), pk2(b.x, b.y), pk2(b.z, b.w) };
  return r;
}

// Load 8 A-fragments (16 rows x 256 K, bf16) from the per-wave LDS tile.
// ISA 16-bit A 16x32 layout: lane = M%16 ; lane-half hh selects K sub-block:
//   q[0] <- K = kb*32 + hh*8 + [0..7],  q[1] <- K = kb*32 + 16 + hh*8 + [0..7]
__device__ __forceinline__ void load_A_lds(Frag* A, const unsigned short* hbuf,
                                           int r, int hh) {
#pragma unroll
  for (int kb = 0; kb < 8; ++kb) {
    const unsigned short* p = hbuf + r * 256 + kb * 32 + hh * 8;
    A[kb].q[0] = *(const v4u*)(p);
    A[kb].q[1] = *(const v4u*)(p + 16);
  }
}

// One 256->256 layer: D = relu(A * W + b), result written as bf16 into hbuf.
// W is bf16 row-major [K=256][N=256].  B layout: lane%16 = N col; elements
// K = hh*8 + [0..7], 16 + hh*8 + [0..7] of the 32-K tile (column gather).
__device__ __forceinline__ void ffn_layer(const Frag* A,
                                          const unsigned short* __restrict__ W,
                                          const float* __restrict__ bias,
                                          unsigned short* hbuf,
                                          int lane, int r, int hh) {
  const v8f vzero = {0.f, 0.f, 0.f, 0.f, 0.f, 0.f, 0.f, 0.f};
#pragma unroll 1
  for (int nc = 0; nc < 4; ++nc) {           // N chunks of 64
    const int nb0 = nc * 64;
    v8f acc0 = vzero, acc1 = vzero, acc2 = vzero, acc3 = vzero;
#pragma unroll
    for (int kb = 0; kb < 8; ++kb) {         // K steps of 32
      const unsigned short* wcol = W + (size_t)(kb * 32 + hh * 8) * 256 + nb0 + r;
      Frag B0, B1, B2, B3;
#pragma unroll
      for (int d = 0; d < 8; ++d) {
        const unsigned short* p = wcol + (size_t)((d < 4) ? 2 * d : 2 * d + 8) * 256;
        B0.q[d >> 2][d & 3] = (unsigned int)p[0]  | ((unsigned int)p[256] << 16);
        B1.q[d >> 2][d & 3] = (unsigned int)p[16] | ((unsigned int)p[272] << 16);
        B2.q[d >> 2][d & 3] = (unsigned int)p[32] | ((unsigned int)p[288] << 16);
        B3.q[d >> 2][d & 3] = (unsigned int)p[48] | ((unsigned int)p[304] << 16);
      }
      acc0 = __builtin_amdgcn_wmma_f32_16x16x32_bf16(false, A[kb].bf, false, B0.bf,
                                                     (short)0, acc0, false, false);
      acc1 = __builtin_amdgcn_wmma_f32_16x16x32_bf16(false, A[kb].bf, false, B1.bf,
                                                     (short)0, acc1, false, false);
      acc2 = __builtin_amdgcn_wmma_f32_16x16x32_bf16(false, A[kb].bf, false, B2.bf,
                                                     (short)0, acc2, false, false);
      acc3 = __builtin_amdgcn_wmma_f32_16x16x32_bf16(false, A[kb].bf, false, B3.bf,
                                                     (short)0, acc3, false, false);
    }
    // Epilogue: D tile layout VGPR p -> M = p + 8*hh, N = nb + r
    auto epi = [&](v8f acc, int nb) {
      float b = bias[nb + r];
#pragma unroll
      for (int p = 0; p < 8; ++p) {
        int row = p + hh * 8;
        float v = fmaxf(acc[p] + b, 0.f);
        hbuf[row * 256 + nb + r] = f2bf(v);
      }
    };
    epi(acc0, nb0); epi(acc1, nb0 + 16); epi(acc2, nb0 + 32); epi(acc3, nb0 + 48);
  }
}

#define WAVES 8

__global__ __launch_bounds__(256) void mlp_topk_kernel(
    const float* __restrict__ X,
    const unsigned short* __restrict__ W1,
    const float* __restrict__ b1,
    const unsigned short* __restrict__ W2,
    const float* __restrict__ b2,
    const unsigned short* __restrict__ W3,   // padded to [256][32]
    const float* __restrict__ b3,
    float* __restrict__ Out) {
  __shared__ unsigned short smem[WAVES][16 * 256];   // 8 KB activation tile / wave

  const int tid  = threadIdx.x;
  const int wave = tid >> 5;
  const int lane = tid & 31;
  const int r    = lane & 15;   // M row within tile (A) / N col (B,C,D)
  const int hh   = lane >> 4;   // lane half
  const int gRow = (blockIdx.x * WAVES + wave) * 16;
  unsigned short* hbuf = &smem[wave][0];

  Frag A[8];

  // ---- Layer 1: A fragments straight from global fp32 X, converted to bf16
  {
    const float* xrow = X + (size_t)(gRow + r) * 256;
#pragma unroll
    for (int kb = 0; kb < 8; ++kb) {
      const float* p0 = xrow + kb * 32 + hh * 8;
      v4f f0 = *(const v4f*)(p0);
      v4f f1 = *(const v4f*)(p0 + 4);
      v4f f2 = *(const v4f*)(p0 + 16);
      v4f f3 = *(const v4f*)(p0 + 20);
      A[kb].q[0] = pack8(f0, f1);
      A[kb].q[1] = pack8(f2, f3);
    }
  }
  ffn_layer(A, W1, b1, hbuf, lane, r, hh);
  asm volatile("s_wait_dscnt 0" ::: "memory");

  // ---- Layer 2
  load_A_lds(A, hbuf, r, hh);
  ffn_layer(A, W2, b2, hbuf, lane, r, hh);
  asm volatile("s_wait_dscnt 0" ::: "memory");

  // ---- Layer 3: 256 -> 24 (padded to 32), no relu
  load_A_lds(A, hbuf, r, hh);
  {
    const v8f vzero = {0.f, 0.f, 0.f, 0.f, 0.f, 0.f, 0.f, 0.f};
    v8f acc0 = vzero, acc1 = vzero;
#pragma unroll
    for (int kb = 0; kb < 8; ++kb) {
      const unsigned short* wcol = W3 + (size_t)(kb * 32 + hh * 8) * 32 + r;
      Frag B0, B1;
#pragma unroll
      for (int d = 0; d < 8; ++d) {
        const unsigned short* p = wcol + ((d < 4) ? 2 * d : 2 * d + 8) * 32;
        B0.q[d >> 2][d & 3] = (unsigned int)p[0]  | ((unsigned int)p[32] << 16);
        B1.q[d >> 2][d & 3] = (unsigned int)p[16] | ((unsigned int)p[48] << 16);
      }
      acc0 = __builtin_amdgcn_wmma_f32_16x16x32_bf16(false, A[kb].bf, false, B0.bf,
                                                     (short)0, acc0, false, false);
      acc1 = __builtin_amdgcn_wmma_f32_16x16x32_bf16(false, A[kb].bf, false, B1.bf,
                                                     (short)0, acc1, false, false);
    }
    asm volatile("s_wait_dscnt 0" ::: "memory");   // A-frag loads done; hbuf reusable
    float* lf = (float*)hbuf;                      // logits: 16 rows x 32 (24 valid)
    {
      float bb0 = b3[r];
      int   col1 = 16 + r;
      float bb1 = (col1 < 24) ? b3[col1] : 0.f;
#pragma unroll
      for (int p = 0; p < 8; ++p) {
        int row = p + hh * 8;
        lf[row * 32 + r]    = acc0[p] + bb0;
        lf[row * 32 + col1] = acc1[p] + bb1;
      }
    }
    asm volatile("s_wait_dscnt 0" ::: "memory");

    // ---- Per-row softmax over SEL (all but cols 0,12), rank-based top-6,
    //      straight-through value (hard - soft) + soft.  One row per lane 0..15.
    if (lane < 16) {
      float lg[24];
#pragma unroll
      for (int j = 0; j < 24; ++j) lg[j] = lf[r * 32 + j];
      float sel[22];
      {
        int c = 0;
#pragma unroll
        for (int j = 0; j < 24; ++j)
          if (j != 0 && j != 12) sel[c++] = lg[j];
      }
      float m = sel[0];
#pragma unroll
      for (int i = 1; i < 22; ++i) m = fmaxf(m, sel[i]);
      float e[22];
      float s = 0.f;
#pragma unroll
      for (int i = 0; i < 22; ++i) { e[i] = __expf(sel[i] - m); s += e[i]; }
      float inv = 1.0f / s;

      float outv[24];
      outv[0] = 1.0f; outv[12] = 1.0f;
      {
        int c = 0;
#pragma unroll
        for (int j = 0; j < 24; ++j) {
          if (j == 0 || j == 12) continue;
          int rank = 0;   // ties broken toward smaller index (matches top_k)
#pragma unroll
          for (int i = 0; i < 22; ++i)
            rank += (sel[i] > sel[c]) || ((sel[i] == sel[c]) && (i < c));
          float hd = (rank < 6) ? 1.0f : 0.0f;
          float so = e[c] * inv;
          outv[j] = (hd - so) + so;
          ++c;
        }
      }
#pragma unroll
      for (int j = 0; j < 24; ++j) lf[512 + r * 24 + j] = outv[j];
    }
    asm volatile("s_wait_dscnt 0" ::: "memory");

    // Coalesced b128 store of this wave's 16x24 outputs
    const v4f* src = (const v4f*)(lf + 512);
    v4f*       dst = (v4f*)(Out + (size_t)gRow * 24);
#pragma unroll
    for (int i = 0; i < 3; ++i) dst[lane + i * 32] = src[lane + i * 32];
  }
}

// ---------------------------------------------------------------------------
// Prep: convert W1/W2 to bf16, W3 to bf16 padded [256][32] (cols 24..31 = 0).
// ---------------------------------------------------------------------------
__global__ __launch_bounds__(256) void cvt_weights_kernel(
    const float* __restrict__ W1f, const float* __restrict__ W2f,
    const float* __restrict__ W3f,
    unsigned short* __restrict__ W1b, unsigned short* __restrict__ W2b,
    unsigned short* __restrict__ W3b) {
  int tid = blockIdx.x * blockDim.x + threadIdx.x;
  if (tid < 256 * 256) {
    W1b[tid] = f2bf(W1f[tid]);
    W2b[tid] = f2bf(W2f[tid]);
  }
  if (tid < 256 * 32) {
    int row = tid >> 5, col = tid & 31;
    W3b[tid] = (col < 24) ? f2bf(W3f[row * 24 + col]) : (unsigned short)0;
  }
}

extern "C" void kernel_launch(void* const* d_in, const int* in_sizes, int n_in,
                              void* d_out, int out_size, void* d_ws, size_t ws_size,
                              hipStream_t stream) {
  const float* X  = (const float*)d_in[0];
  const float* W1 = (const float*)d_in[1];
  const float* b1 = (const float*)d_in[2];
  const float* W2 = (const float*)d_in[3];
  const float* b2 = (const float*)d_in[4];
  const float* W3 = (const float*)d_in[5];
  const float* b3 = (const float*)d_in[6];

  unsigned short* W1b = (unsigned short*)d_ws;         // 256*256
  unsigned short* W2b = W1b + 256 * 256;               // 256*256
  unsigned short* W3b = W2b + 256 * 256;               // 256*32 (padded)

  cvt_weights_kernel<<<256, 256, 0, stream>>>(W1, W2, W3, W1b, W2b, W3b);

  const int Brows  = in_sizes[0] / 256;                // 131072
  const int blocks = Brows / (WAVES * 16);             // 128 rows / block
  mlp_topk_kernel<<<blocks, 256, 0, stream>>>(X, W1b, b1, W2b, b2, W3b, b3,
                                              (float*)d_out);
}
